// BiLSTM_20633022890537
// MI455X (gfx1250) — matrix operand order repaired and checked by reading, not compiled, not run on hardware
//
#include <hip/hip_runtime.h>

// BiLSTM for MI455X (gfx1250): bf16 WMMA GEMMs + persistent recurrent kernel
// with W_hi resident in LDS (async-filled). K loops fully unrolled so the
// scheduler pipelines loads ahead of WMMAs without register-rotation moves.
// B=32, S=512, I=H=1024, 4H=4096.

#define SLEN 512
#define BATCH 32
#define IDIM 1024
#define HDIM 1024
#define G4 4096
#define WELEMS ((size_t)4096 * 1024)   // elements per weight matrix (4H x K)
#define LDS_STRIDE 1032                // 1024 + 8 pad: 516 words % 64 banks = 4

typedef __bf16 bf16x8  __attribute__((ext_vector_type(8)));
typedef __bf16 bf16x16 __attribute__((ext_vector_type(16)));
typedef float  floatx8 __attribute__((ext_vector_type(8)));
typedef int    int4v   __attribute__((ext_vector_type(4)));

#ifndef __has_builtin
#define __has_builtin(x) 0
#endif
#if __has_builtin(__builtin_amdgcn_global_load_async_to_lds_b128)
#define HAVE_ASYNC_LDS 1
#else
#define HAVE_ASYNC_LDS 0
#endif

// Builtin signature (from hipcc diagnostic): param0 = int4 addrspace(1)*,
// param1 = int4 addrspace(3)*, then imm offset, imm cpol.
typedef __attribute__((address_space(1))) int4v* as1_i4p;
typedef __attribute__((address_space(3))) int4v* as3_i4p;

static __device__ __forceinline__ unsigned short f2bf(float f) {
  unsigned u = __float_as_uint(f);
  u += 0x7FFFu + ((u >> 16) & 1u);     // round-to-nearest-even
  return (unsigned short)(u >> 16);
}

static __device__ __forceinline__ float sigf(float x) {
  return 1.0f / (1.0f + __expf(-x));
}

#define SHUF16(lo, hi) __builtin_shufflevector(lo, hi, 0, 1, 2, 3, 4, 5, 6, 7, \
                                               8, 9, 10, 11, 12, 13, 14, 15)

// ---------------------------------------------------------------------------
// Kernel 0: fp32 -> bf16 conversion of x and the four weight matrices,
// plus zeroing of the grid-sync counters. Runs fresh every launch.
// ---------------------------------------------------------------------------
__global__ void __launch_bounds__(256) k_convert(
    const float* __restrict__ x,
    const float* __restrict__ wii, const float* __restrict__ whi,
    const float* __restrict__ wii_r, const float* __restrict__ whi_r,
    unsigned short* __restrict__ xb, unsigned short* __restrict__ wb,
    unsigned int* __restrict__ cnt) {
  const size_t NX = (size_t)BATCH * SLEN * IDIM;   // 16.77M
  size_t i = (size_t)blockIdx.x * blockDim.x + threadIdx.x;
  if (i < NX) xb[i] = f2bf(x[i]);
  if (i < WELEMS) {
    wb[i]              = f2bf(wii[i]);
    wb[i + WELEMS]     = f2bf(whi[i]);
    wb[i + 2 * WELEMS] = f2bf(wii_r[i]);
    wb[i + 3 * WELEMS] = f2bf(whi_r[i]);
  }
  if (i < 64) cnt[i] = 0;
}

// ---------------------------------------------------------------------------
// Kernel 1: xp[dir][t][b][g] = x[b][s] . W_ii[dir][g] + b_i[dir][g]   (fp32)
// s = t (dir 0) or S-1-t (dir 1). One wave = 1 M-tile x 8 N-tiles: 8
// independent WMMA chains per A fragment (8x A reuse), fully unrolled K loop
// so loads hoist ahead of their consuming WMMAs with partial waits and fresh
// registers per iteration (no rotation moves).
// ---------------------------------------------------------------------------
__global__ void __launch_bounds__(256) k_xproj(
    const unsigned short* __restrict__ xb,
    const unsigned short* __restrict__ wb,
    const float* __restrict__ bias_f,
    const float* __restrict__ bias_r,
    float* __restrict__ xp) {
  const int lane  = threadIdx.x & 31;
  const int w     = threadIdx.x >> 5;
  const int gw    = blockIdx.x * 8 + w;   // 65536 waves total
  const int dir   = gw >> 15;
  const int rem   = gw & 32767;
  const int mt    = rem >> 5;             // 0..1023  (M-tile over (t,b))
  const int ng    = rem & 31;             // 0..31    (group of 8 N-tiles)
  const int n0    = ng * 128;

  const int lhalf = lane >> 4;            // 0/1
  const int lrow  = lane & 15;
  const int hi8   = lhalf * 8;

  const int m = mt * 16 + lrow;           // global M row = t*32 + b
  const int t = m >> 5;                   // constant within tile (16 | 32)
  const int b = m & 31;
  const int s = dir ? (SLEN - 1 - t) : t;
  const unsigned short* arow = xb + ((size_t)b * SLEN + s) * IDIM + lhalf * 16;
  const unsigned short* wrow0 =
      wb + (size_t)(dir * 2) * WELEMS + (size_t)(n0 + lrow) * IDIM + lhalf * 16;
  const float* bias = dir ? bias_r : bias_f;

  floatx8 acc[8] = {};

#pragma unroll
  for (int k = 0; k < IDIM; k += 32) {
    const bf16x8 alo = *(const bf16x8*)(arow + k);
    const bf16x8 ahi = *(const bf16x8*)(arow + k + 8);
    const bf16x16 a = SHUF16(alo, ahi);
#pragma unroll
    for (int j = 0; j < 8; ++j) {
      const bf16x16 bf =
          *(const bf16x16*)(wrow0 + (size_t)j * 16 * IDIM + k);
      acc[j] = __builtin_amdgcn_wmma_f32_16x16x32_bf16(
          false, a, false, bf, (short)0, acc[j], false, false);
    }
  }

  // Store C (+bias): lane holds column g, rows hi8+r.
  const size_t rowbase = (size_t)(dir * SLEN + t) * 32 + (mt & 1) * 16 + hi8;
#pragma unroll
  for (int j = 0; j < 8; ++j) {
    const int g = n0 + j * 16 + lrow;
    const float bv = bias[g];
#pragma unroll
    for (int r = 0; r < 8; ++r) {
      xp[(rowbase + r) * (size_t)G4 + g] = acc[j][r] + bv;
    }
  }
}

// ---------------------------------------------------------------------------
// Kernel 2: persistent recurrent scan. 64 blocks x 128 threads:
// blocks 0-31 = forward, 32-63 = backward. Block owns 32 hidden units and
// stages its W_hi slice (4 gates x 32 cols x 1024 K bf16 = 256 KB) in LDS
// once via GLOBAL_LOAD_ASYNC_TO_LDS_B128. Wave (mt,jt) owns batch tile mt and
// 16 hidden columns, computes all 4 gate tiles so c-state stays in registers
// for all 512 steps. The K loop is fully unrolled; A (h_prev) global loads
// and LDS B fragments get hoisted ahead of their WMMAs by the scheduler.
// xp for the step is prefetched before the inter-block barrier.
// ---------------------------------------------------------------------------
__global__ void __launch_bounds__(128) k_recurrent(
    const unsigned short* __restrict__ wb,
    const float* __restrict__ xp,
    unsigned short* __restrict__ hbuf,   // [dir][phase][B][H] bf16 ping-pong
    float* __restrict__ out,             // (B, S, 2H) fp32
    unsigned int* __restrict__ cnt) {
  extern __shared__ char smem[];
  unsigned short* sh = (unsigned short*)smem;  // [128 rows][LDS_STRIDE]

  const int dir   = blockIdx.x >> 5;
  const int blk   = blockIdx.x & 31;
  const int tid   = threadIdx.x;
  const int lane  = tid & 31;
  const int w     = tid >> 5;
  const int mt    = w & 1;               // batch tile: rows 16*mt..16*mt+15
  const int jt    = w >> 1;              // 0..1
  const int lhalf = lane >> 4;
  const int lrow  = lane & 15;
  const int hi8   = lhalf * 8;
  const int jloc  = jt * 16 + lrow;      // 0..31 within block slice
  const int jh    = blk * 32 + jloc;     // global hidden unit column
  const unsigned short* whi = wb + (size_t)(dir * 2 + 1) * WELEMS;
  unsigned int* mycnt = cnt + dir * 32;

  // ---- stage W_hi slice into LDS: row r = gate*32 + jloc_w ----
  for (int c = tid; c < 128 * 128; c += 128) {   // 16384 x 16B chunks
    const int row  = c >> 7;
    const int ke   = (c & 127) << 3;             // element offset, 8 el / 16B
    const int wrow = (row >> 5) * HDIM + blk * 32 + (row & 31);
    const unsigned short* g = whi + (size_t)wrow * HDIM + ke;
    unsigned short* l = sh + row * LDS_STRIDE + ke;
#if HAVE_ASYNC_LDS
    __builtin_amdgcn_global_load_async_to_lds_b128((as1_i4p)(void*)g,
                                                   (as3_i4p)(void*)l, 0, 0);
#else
    *(uint4*)l = *(const uint4*)g;
#endif
  }
#if HAVE_ASYNC_LDS
#if __has_builtin(__builtin_amdgcn_s_wait_asynccnt)
  __builtin_amdgcn_s_wait_asynccnt(0);
#else
  asm volatile("s_wait_asynccnt 0x0" ::: "memory");
#endif
#endif
  __syncthreads();

  // Per-gate LDS row base for this wave's B fragments.
  const unsigned short* brow[4];
#pragma unroll
  for (int g4 = 0; g4 < 4; ++g4)
    brow[g4] = sh + (size_t)(g4 * 32 + jloc) * LDS_STRIDE + lhalf * 16;

  const int marow = mt * 16 + lrow;      // A row (batch) for this lane

  float cst[8];
#pragma unroll
  for (int r = 0; r < 8; ++r) cst[r] = 0.0f;

  for (int t = 0; t < SLEN; ++t) {
    // Prefetch this step's gate pre-activations (independent of the barrier,
    // so their L2 latency hides behind the spin + GEMM).
    const size_t xrow0 = (size_t)(dir * SLEN + t) * 32;
    float xg[4][8];
#pragma unroll
    for (int g4 = 0; g4 < 4; ++g4)
#pragma unroll
      for (int r = 0; r < 8; ++r) {
        const int b = mt * 16 + hi8 + r;
        xg[g4][r] = xp[(xrow0 + b) * (size_t)G4 + g4 * 1024 + jh];
      }

    if (t > 0) {
      if (tid == 0) {
        const unsigned target = (unsigned)t * 32u;
        while (__hip_atomic_load(mycnt, __ATOMIC_ACQUIRE,
                                 __HIP_MEMORY_SCOPE_AGENT) < target) {
          __builtin_amdgcn_s_sleep(1);
        }
      }
      __syncthreads();
      __threadfence();
    }

    floatx8 acc[4] = {};

    if (t > 0) {
      const unsigned short* hprev =
          hbuf + (size_t)(dir * 2 + ((t - 1) & 1)) * BATCH * HDIM;
      const unsigned short* arow =
          hprev + (size_t)marow * HDIM + lhalf * 16;
#pragma unroll
      for (int k = 0; k < HDIM; k += 32) {
        const bf16x8 alo = *(const bf16x8*)(arow + k);
        const bf16x8 ahi = *(const bf16x8*)(arow + k + 8);
        const bf16x16 a = SHUF16(alo, ahi);
#pragma unroll
        for (int g4 = 0; g4 < 4; ++g4) {
          const bf16x16 bf = *(const bf16x16*)(brow[g4] + k);
          acc[g4] = __builtin_amdgcn_wmma_f32_16x16x32_bf16(
              false, a, false, bf, (short)0, acc[g4], false, false);
        }
      }
    }

    // Pointwise gates + cell update (c persistent in registers).
    const int s_out = dir ? (SLEN - 1 - t) : t;
    unsigned short* hcur =
        hbuf + (size_t)(dir * 2 + (t & 1)) * BATCH * HDIM;
#pragma unroll
    for (int r = 0; r < 8; ++r) {
      const int b = mt * 16 + hi8 + r;
      const float gi = acc[0][r] + xg[0][r];
      const float gf = acc[1][r] + xg[1][r];
      const float gg = acc[2][r] + xg[2][r];
      const float go = acc[3][r] + xg[3][r];
      const float c  = sigf(gf) * cst[r] + sigf(gi) * tanhf(gg);
      const float h  = sigf(go) * tanhf(c);
      cst[r] = c;
      hcur[(size_t)b * HDIM + jh] = f2bf(h);
      out[((size_t)b * SLEN + s_out) * 2048 + dir * 1024 + jh] = h;
    }

    __threadfence();
    __syncthreads();
    if (tid == 0) {
      __hip_atomic_fetch_add(mycnt, 1u, __ATOMIC_RELEASE,
                             __HIP_MEMORY_SCOPE_AGENT);
    }
  }
}

// ---------------------------------------------------------------------------
// Launch. Workspace layout (bytes, 256-aligned):
//   [0, 33.5M)        xb  : x as bf16
//   [33.5M, 67.1M)    wb  : W_ii, W_hi, W_ii_r, W_hi_r as bf16
//   [67.1M, 604.0M)   xp  : (2, S, B, 4H) fp32 pre-activations
//   [604.0M, +256K)   hbuf: bf16 h ping-pong, 2 dirs x 2 phases
//   [+256K, +256)     cnt : sync counters
// ---------------------------------------------------------------------------
extern "C" void kernel_launch(void* const* d_in, const int* in_sizes, int n_in,
                              void* d_out, int out_size, void* d_ws, size_t ws_size,
                              hipStream_t stream) {
  const float* x      = (const float*)d_in[0];
  const float* W_ii   = (const float*)d_in[1];
  const float* W_hi   = (const float*)d_in[2];
  const float* b_i    = (const float*)d_in[3];
  const float* W_ii_r = (const float*)d_in[4];
  const float* W_hi_r = (const float*)d_in[5];
  const float* b_i_r  = (const float*)d_in[6];
  float* out = (float*)d_out;

  char* ws = (char*)d_ws;
  const size_t off_xb  = 0;
  const size_t off_wb  = off_xb + (size_t)BATCH * SLEN * IDIM * 2;        // 33.5 MB
  const size_t off_xp  = off_wb + 4 * WELEMS * 2;                         // +33.5 MB
  const size_t off_h   = off_xp + (size_t)2 * SLEN * BATCH * G4 * 4;      // +536.9 MB
  const size_t off_cnt = off_h + (size_t)2 * 2 * BATCH * HDIM * 2;        // +256 KB

  unsigned short* xb   = (unsigned short*)(ws + off_xb);
  unsigned short* wb   = (unsigned short*)(ws + off_wb);
  float*          xp   = (float*)(ws + off_xp);
  unsigned short* hbuf = (unsigned short*)(ws + off_h);
  unsigned int*   cnt  = (unsigned int*)(ws + off_cnt);

  const size_t lds_bytes = (size_t)128 * LDS_STRIDE * 2;   // 264,192 B

  // Opt in to >64KB dynamic LDS (host-side attribute, not a stream op).
  static int lds_attr_set = 0;
  if (!lds_attr_set) {
    (void)hipFuncSetAttribute((const void*)k_recurrent,
                              hipFuncAttributeMaxDynamicSharedMemorySize,
                              (int)lds_bytes);
    lds_attr_set = 1;
  }

  // Kernel 0: conversions + counter reset (16.77M elems -> 65536 blocks).
  k_convert<<<65536, 256, 0, stream>>>(x, W_ii, W_hi, W_ii_r, W_hi_r,
                                       xb, wb, cnt);

  // Kernel 1: input projections, 65536 waves = 8192 blocks of 8 waves.
  k_xproj<<<8192, 256, 0, stream>>>(xb, wb, b_i, b_i_r, xp);

  // Kernel 2: persistent recurrent scan, 32 blocks per direction.
  k_recurrent<<<64, 128, lds_bytes, stream>>>(wb, xp, hbuf, out, cnt);
}
